// DroplessGroupedGEMM_32255204393502
// MI455X (gfx1250) — matrix operand order repaired and checked by reading, not compiled, not run on hardware
//
#include <hip/hip_runtime.h>
#include <hip/hip_bf16.h>

// Problem constants (match reference)
#define E   8
#define C   768
#define H   1024
#define F   4096
#define F2  (2*F)
#define T   4096
#define EC  (E*C)

typedef __attribute__((ext_vector_type(16))) __bf16        v16bf;
typedef __attribute__((ext_vector_type(8)))  float         v8f;
typedef __attribute__((ext_vector_type(8)))  unsigned int  v8u;
typedef __attribute__((ext_vector_type(4)))  unsigned int  v4u;
typedef __attribute__((ext_vector_type(4)))  float         v4f;

// ---------- helpers ----------

// fp32 -> bf16 round-to-nearest-even (bit trick)
__device__ __forceinline__ unsigned short f2bf(float f) {
    unsigned int u = __float_as_uint(f);
    u += ((u >> 16) & 1u) + 0x7FFFu;
    return (unsigned short)(u >> 16);
}

// pack 8 floats into 16 bytes of bf16
__device__ __forceinline__ v4u pack8(const float* v) {
    v4u q;
    q.x = (unsigned)f2bf(v[0]) | ((unsigned)f2bf(v[1]) << 16);
    q.y = (unsigned)f2bf(v[2]) | ((unsigned)f2bf(v[3]) << 16);
    q.z = (unsigned)f2bf(v[4]) | ((unsigned)f2bf(v[5]) << 16);
    q.w = (unsigned)f2bf(v[6]) | ((unsigned)f2bf(v[7]) << 16);
    return q;
}

// load a 32-byte WMMA fragment (16 bf16 per lane) -> two global_load_b128
__device__ __forceinline__ v16bf load_frag(const char* p) {
    v8u q = *(const v8u*)p;
    return __builtin_bit_cast(v16bf, q);
}

__device__ __forceinline__ v8f wmma_bf16(v16bf a, v16bf b, v8f c) {
    return __builtin_amdgcn_wmma_f32_16x16x32_bf16(
        /*neg_a=*/false, a, /*neg_b=*/false, b,
        /*c_mod=*/(short)0, c, /*reuse_a=*/false, /*reuse_b=*/false);
}

// Packed-layout byte offsets (per ISA 7.12.2 VGPR layouts, wave32):
// A fragment (16x32 bf16): lane L = h*16+m holds row m, K = {kT*32 + c*16 + h*8 + j}
//   offset(mT,kT,L) = ((mT*KT + kT)*32 + L)*32  (+ c*16 for chunk c)
// B fragment (32x16 bf16): lane L = h*16+n holds col n, K = {kT*32 + c*16 + h*8 + j}
//   offset(nT,kT,L) = ((nT*KT + kT)*32 + L)*32  (+ c*16)

// ---------- kernel 0: FCFS slot assignment ----------
__global__ void moe_assign_kernel(const int* __restrict__ eids,
                                  int* __restrict__ slot,
                                  float* __restrict__ valid_out) {
    int e = threadIdx.x;
    if (e >= E) return;
    int cnt = 0;
    for (int t = 0; t < T; ++t) {
        int id = eids[t];
        id = id < 0 ? 0 : (id >= E ? E - 1 : id);
        if (id == e) {
            if (cnt < C) { slot[t] = e * C + cnt; valid_out[t] = 1.0f; }
            else         { slot[t] = -1;          valid_out[t] = 0.0f; }
            ++cnt;
        }
    }
}

// ---------- kernel 1: pack w_gate_up [E,H,2F] fp32 -> bf16 B-fragments ----------
// fp32 source is read exactly once: non-temporal so it doesn't evict the
// packed bf16 weights (~201 MB ~ L2) that the GEMMs re-read.
__global__ __launch_bounds__(256)
void moe_pack_wgu_kernel(const float* __restrict__ w, char* __restrict__ dst) {
    size_t gid = (size_t)blockIdx.x * 256 + threadIdx.x;   // E*512*32*32 = 2^22
    int lane = (int)(gid & 31);
    int kT   = (int)((gid >> 5) & 31);     // H/32
    int nT   = (int)((gid >> 10) & 511);   // 2F/16
    int e    = (int)(gid >> 19);
    int h = lane >> 4, n = lane & 15;
    float v[16];
#pragma unroll
    for (int c = 0; c < 2; ++c)
#pragma unroll
        for (int j = 0; j < 8; ++j) {
            int k = kT * 32 + c * 16 + h * 8 + j;
            v[c * 8 + j] = __builtin_nontemporal_load(
                &w[((size_t)e * H + k) * F2 + (size_t)nT * 16 + n]);
        }
    v4u* d = (v4u*)(dst + ((size_t)((e * 512 + nT) * 32 + kT) * 32 + lane) * 32);
    d[0] = pack8(v); d[1] = pack8(v + 8);
}

// ---------- kernel 2: pack w_down [E,F,H] fp32 -> bf16 B-fragments ----------
__global__ __launch_bounds__(256)
void moe_pack_wd_kernel(const float* __restrict__ w, char* __restrict__ dst) {
    size_t gid = (size_t)blockIdx.x * 256 + threadIdx.x;   // E*64*128*32 = 2^21
    int lane = (int)(gid & 31);
    int kT   = (int)((gid >> 5) & 127);    // F/32
    int nT   = (int)((gid >> 12) & 63);    // H/16
    int e    = (int)(gid >> 18);
    int h = lane >> 4, n = lane & 15;
    float v[16];
#pragma unroll
    for (int c = 0; c < 2; ++c)
#pragma unroll
        for (int j = 0; j < 8; ++j) {
            int k = kT * 32 + c * 16 + h * 8 + j;
            v[c * 8 + j] = __builtin_nontemporal_load(
                &w[((size_t)e * F + k) * H + (size_t)nT * 16 + n]);
        }
    v4u* d = (v4u*)(dst + ((size_t)((e * 64 + nT) * 128 + kT) * 32 + lane) * 32);
    d[0] = pack8(v); d[1] = pack8(v + 8);
}

// ---------- kernel 3: dispatch tokens -> packed bf16 A-fragments ----------
__global__ void moe_dispatch_kernel(const float* __restrict__ tokens,
                                    const int* __restrict__ slot,
                                    char* __restrict__ xA) {
    int t = blockIdx.x;
    int s = slot[t];
    if (s < 0) return;                      // dropped token (uniform per block)
    int j  = threadIdx.x;                   // 64 threads: (kT, h)
    int kT = j >> 1, h = j & 1;
    int mT = s >> 4, m = s & 15;
    int L  = h * 16 + m;
    const float* tok = tokens + (size_t)t * H;
    float v0[8], v1[8];
#pragma unroll
    for (int jj = 0; jj < 8; ++jj) {
        v0[jj] = __builtin_nontemporal_load(&tok[kT * 32 + h * 8 + jj]);       // c=0
        v1[jj] = __builtin_nontemporal_load(&tok[kT * 32 + 16 + h * 8 + jj]);  // c=1
    }
    v4u* d = (v4u*)(xA + ((size_t)(mT * (H / 32) + kT) * 32 + L) * 32);
    d[0] = pack8(v0); d[1] = pack8(v1);
}

// ---------- kernel 4: grouped GEMM1 + fused SwiGLU, emits packed bf16 hA ----------
// Block = 128 rows x 64 gate cols (+ paired up cols). 8 waves; each wave owns
// 2 mTiles x 2 nTile-cells with gate+up accumulators:
//   8 WMMAs per K-step from 6 fragment loads (2 A + 4 B).
__global__ __launch_bounds__(256)
void moe_gemm1_swiglu_kernel(const char* __restrict__ xA,
                             const char* __restrict__ wguB,
                             char* __restrict__ hA) {
    __shared__ float stage[8][16 * 17];     // per-wave D->A transpose tile
    int w = threadIdx.x >> 5, lane = threadIdx.x & 31;
    int mT0   = blockIdx.y * 8 + (w & 3) * 2;     // global row tile (0..EC/16-1)
    int e     = (blockIdx.y * 128) / C;           // expert of this row block
    int nBase = blockIdx.x * 4 + (w >> 2) * 2;    // gate col tile (0..255)
    const char* bE  = wguB + (size_t)e * ((size_t)512 * 32 * 32 * 32);
    const char* aP0 = xA + ((size_t)(mT0 + 0) * (H / 32) * 32 + lane) * 32;
    const char* aP1 = xA + ((size_t)(mT0 + 1) * (H / 32) * 32 + lane) * 32;

    v8f accG[2][2] = {}, accU[2][2] = {};
    for (int kT = 0; kT < H / 32; ++kT) {
        v16bf a0  = load_frag(aP0 + (size_t)kT * 1024);
        v16bf a1  = load_frag(aP1 + (size_t)kT * 1024);
        v16bf bg0 = load_frag(bE + ((size_t)((nBase + 0)   * 32 + kT) * 32 + lane) * 32);
        v16bf bg1 = load_frag(bE + ((size_t)((nBase + 1)   * 32 + kT) * 32 + lane) * 32);
        v16bf bu0 = load_frag(bE + ((size_t)((nBase + 256) * 32 + kT) * 32 + lane) * 32);
        v16bf bu1 = load_frag(bE + ((size_t)((nBase + 257) * 32 + kT) * 32 + lane) * 32);
        accG[0][0] = wmma_bf16(a0, bg0, accG[0][0]);
        accG[0][1] = wmma_bf16(a0, bg1, accG[0][1]);
        accU[0][0] = wmma_bf16(a0, bu0, accU[0][0]);
        accU[0][1] = wmma_bf16(a0, bu1, accU[0][1]);
        accG[1][0] = wmma_bf16(a1, bg0, accG[1][0]);
        accG[1][1] = wmma_bf16(a1, bg1, accG[1][1]);
        accU[1][0] = wmma_bf16(a1, bu0, accU[1][0]);
        accU[1][1] = wmma_bf16(a1, bu1, accU[1][1]);
    }

    int Mhi = (lane >> 4) * 8, n = lane & 15;
    int m = lane & 15, h2 = lane >> 4;
#pragma unroll
    for (int cc = 0; cc < 4; ++cc) {        // (mi, ni) cells
        int mi = cc >> 1, i = cc & 1;
        v8f g = accG[mi][i];
        v8f u = accU[mi][i];
        // SwiGLU: silu(gate) * up, staged to LDS in D-layout
#pragma unroll
        for (int r = 0; r < 8; ++r) {
            float gg = g[r];
            float s  = gg / (1.0f + __expf(-gg));
            stage[w][(Mhi + r) * 17 + n] = s * u[r];
        }
        __syncthreads();
        // read back in A-fragment layout (row m, 8 consecutive K) and pack bf16
        float rv[8];
#pragma unroll
        for (int jj = 0; jj < 8; ++jj) rv[jj] = stage[w][m * 17 + h2 * 8 + jj];
        int mT  = mT0 + mi;
        int nT  = nBase + i;                 // F-dim column tile of this cell
        int kT2 = nT >> 1, c = nT & 1;       // which 16-K half of hA's 32-K tile
        v4u* d = (v4u*)(hA + ((size_t)(mT * (F / 32) + kT2) * 32 + lane) * 32 + c * 16);
        d[0] = pack8(rv);
        __syncthreads();
    }
}

// ---------- kernel 5: grouped GEMM2 (down proj) -> out_ec fp32 ----------
// Block = 128 rows x 64 cols; each wave 2 mT x 2 nT: 4 WMMAs per 4 loads.
__global__ __launch_bounds__(256)
void moe_gemm2_kernel(const char* __restrict__ hA,
                      const char* __restrict__ wdB,
                      float* __restrict__ out_ec) {
    int w = threadIdx.x >> 5, lane = threadIdx.x & 31;
    int mT0   = blockIdx.y * 8 + (w & 3) * 2;
    int e     = (blockIdx.y * 128) / C;
    int nBase = blockIdx.x * 4 + (w >> 2) * 2;    // H col tile (0..63)
    const char* bE  = wdB + (size_t)e * ((size_t)64 * 128 * 32 * 32);
    const char* aP0 = hA + ((size_t)(mT0 + 0) * (F / 32) * 32 + lane) * 32;
    const char* aP1 = hA + ((size_t)(mT0 + 1) * (F / 32) * 32 + lane) * 32;

    v8f acc[2][2] = {};
    for (int kT = 0; kT < F / 32; ++kT) {
        v16bf a0 = load_frag(aP0 + (size_t)kT * 1024);
        v16bf a1 = load_frag(aP1 + (size_t)kT * 1024);
        v16bf b0 = load_frag(bE + ((size_t)((nBase + 0) * 128 + kT) * 32 + lane) * 32);
        v16bf b1 = load_frag(bE + ((size_t)((nBase + 1) * 128 + kT) * 32 + lane) * 32);
        acc[0][0] = wmma_bf16(a0, b0, acc[0][0]);
        acc[0][1] = wmma_bf16(a0, b1, acc[0][1]);
        acc[1][0] = wmma_bf16(a1, b0, acc[1][0]);
        acc[1][1] = wmma_bf16(a1, b1, acc[1][1]);
    }
    int n = lane & 15, Mhi = (lane >> 4) * 8;
#pragma unroll
    for (int mi = 0; mi < 2; ++mi)
#pragma unroll
        for (int i = 0; i < 2; ++i) {
            v8f a = acc[mi][i];
            int col = (nBase + i) * 16 + n;
#pragma unroll
            for (int r = 0; r < 8; ++r) {
                int row = (mT0 + mi) * 16 + Mhi + r;
                out_ec[(size_t)row * H + col] = a[r];
            }
        }
}

// ---------- kernel 6: combine (gather back to token order) ----------
__global__ void moe_combine_kernel(const int* __restrict__ slot,
                                   const float* __restrict__ out_ec,
                                   float* __restrict__ out) {
    int t = blockIdx.x, i = threadIdx.x;     // 256 threads x float4 = H
    int s = slot[t];
    v4f v = {0.0f, 0.0f, 0.0f, 0.0f};
    if (s >= 0) v = ((const v4f*)(out_ec + (size_t)s * H))[i];
    ((v4f*)(out + (size_t)t * H))[i] = v;
}

// ---------- launcher ----------
extern "C" void kernel_launch(void* const* d_in, const int* in_sizes, int n_in,
                              void* d_out, int out_size, void* d_ws, size_t ws_size,
                              hipStream_t stream) {
    const float* tokens = (const float*)d_in[0];
    const int*   eids   = (const int*)d_in[1];
    const float* wgu    = (const float*)d_in[2];
    const float* wd     = (const float*)d_in[3];
    float* out = (float*)d_out;              // [T*H] outputs then [T] valid

    char* ws = (char*)d_ws;
    size_t off = 0;
    int* slot = (int*)(ws + off);            off += (size_t)T * 4;
    off = (off + 255) & ~(size_t)255;
    char* xA   = ws + off;                   off += (size_t)EC * H * 2;     // 12.6 MB
    char* hA   = ws + off;                   off += (size_t)EC * F * 2;     // 50.3 MB
    char* wguB = ws + off;                   off += (size_t)E * H * F2 * 2; // 134 MB
    char* wdB  = ws + off;                   off += (size_t)E * F * H * 2;  // 67 MB
    float* out_ec = (float*)(ws + off);      off += (size_t)EC * H * 4;     // 25 MB
    (void)ws_size; (void)in_sizes; (void)n_in; (void)out_size;

    // 0) FCFS assignment (order-sensitive -> 1 thread per expert), writes valid
    moe_assign_kernel<<<1, 32, 0, stream>>>(eids, slot, out + (size_t)T * H);
    // 1) zero dispatch buffer (empty slots must contribute zeros)
    hipMemsetAsync(xA, 0, (size_t)EC * H * 2, stream);
    // 2) pack weights fp32 -> bf16 WMMA fragments (L2-resident: ~201 MB)
    moe_pack_wgu_kernel<<<(E * 512 * 32 * 32) / 256, 256, 0, stream>>>(wgu, wguB);
    moe_pack_wd_kernel <<<(E * 64 * 128 * 32) / 256, 256, 0, stream>>>(wd, wdB);
    // 3) scatter tokens into packed bf16 A-fragments
    moe_dispatch_kernel<<<T, 64, 0, stream>>>(tokens, slot, xA);
    // 4) grouped GEMM1 + fused SwiGLU -> packed bf16 hA (128-row blocks)
    moe_gemm1_swiglu_kernel<<<dim3(64, EC / 128), 256, 0, stream>>>(xA, wguB, hA);
    // 5) grouped GEMM2 -> out_ec fp32
    moe_gemm2_kernel<<<dim3(16, EC / 128), 256, 0, stream>>>(hA, wdB, out_ec);
    // 6) gather to token order (zeros for dropped tokens)
    moe_combine_kernel<<<T, 256, 0, stream>>>(slot, out_ec, out);
}
